// STGCNBlock_82626580840720
// MI455X (gfx1250) — compile-verified, matrix operand order checked by reading, not run on hardware
//
#include <hip/hip_runtime.h>
#include <hip/hip_bf16.h>

typedef __attribute__((ext_vector_type(16))) __bf16 v16bf;
typedef __attribute__((ext_vector_type(8)))  float  v8f;

#define NB   32
#define CIN  64
#define TT   256
#define VV   25
#define FF   128
#define TD   256
#define EE   48

// workspace layout (bytes)
#define WS_AHAT 0u          // ushort[32*32]        = 2048
#define WS_WGT  2048u       // ushort[128*64]       = 16384 -> 18432
#define WS_TFB  18432u      // float [32*128]       = 16384 -> 34816
#define WS_WKT  34816u      // ushort[3*128*128]    = 98304 -> 133120
#define WS_H    133120u     // ushort[32*25*256*128] bf16 intermediate h

#if defined(__gfx1250__) && __has_builtin(__builtin_amdgcn_tensor_load_to_lds)
#define USE_TDM 1
#endif

// truncating f32 -> bf16 (precision already bf16-limited; avoids 4-op RNE)
__device__ __forceinline__ unsigned short f2bf(float f) {
  return (unsigned short)(__float_as_uint(f) >> 16);
}
// pack two f32 -> bf16x2 with one v_perm_b32
__device__ __forceinline__ unsigned pk2bf(float lo, float hi) {
  return __builtin_amdgcn_perm(__float_as_uint(hi), __float_as_uint(lo),
                               0x07060302u);
}

union BfTile {
  v16bf v;
  uint4 q[2];
  unsigned p[8];
  unsigned short u[16];
};

// ---------------------------------------------------------------------------
// Kernel A: build normalized adjacency, transpose weights to bf16, time proj
// ---------------------------------------------------------------------------
__global__ __launch_bounds__(256)
void prep_kernel(const int* ei, const float* t_emb, const float* W_gcn,
                 const float* b_gcn, const float* W_time, const float* b_time,
                 const float* W_tcn, unsigned char* ws) {
  __shared__ float sA[32 * 32];
  __shared__ float sDinv[32];
  int tid = threadIdx.x;

  for (int i = tid; i < 1024; i += 256) sA[i] = 0.0f;
  __syncthreads();

  if (tid == 0) {
    for (int i = 0; i < VV; ++i) sDinv[i] = 1.0f;           // self loop
    for (int e = 0; e < EE; ++e) sDinv[ei[EE + e]] += 1.0f; // deg from dst
    for (int i = 0; i < VV; ++i) sDinv[i] = rsqrtf(sDinv[i]);
    for (int e = 0; e < EE; ++e) {
      int s = ei[e], d = ei[EE + e];
      sA[d * 32 + s] += sDinv[s] * sDinv[d];
    }
    for (int i = 0; i < VV; ++i) sA[i * 32 + i] += sDinv[i] * sDinv[i];
  }
  __syncthreads();

  unsigned short* Ahat = (unsigned short*)(ws + WS_AHAT);
  unsigned short* WgT  = (unsigned short*)(ws + WS_WGT);
  float*          tfb  = (float*)(ws + WS_TFB);
  unsigned short* WkT  = (unsigned short*)(ws + WS_WKT);

  for (int i = tid; i < 1024; i += 256) Ahat[i] = f2bf(sA[i]);

  // W_gcn^T : [f][c]
  for (int i = tid; i < FF * CIN; i += 256) {
    int f = i >> 6, c = i & 63;
    WgT[i] = f2bf(W_gcn[c * FF + f]);
  }
  // W_tcn per-tap transposed: WkT[k][fo][fi] = W_tcn[fo][fi][k]
  for (int i = tid; i < 3 * FF * FF; i += 256) {
    int k = i / (FF * FF);
    int rem = i - k * FF * FF;
    int fo = rem >> 7, fi = rem & 127;
    WkT[i] = f2bf(W_tcn[(fo * FF + fi) * 3 + k]);
  }
  // tfb[n][f] = t_emb @ W_time + b_time + b_gcn
  for (int o = tid; o < NB * FF; o += 256) {
    int n = o >> 7, f = o & 127;
    float acc = b_time[f] + b_gcn[f];
    for (int td = 0; td < TD; ++td)
      acc += t_emb[n * TD + td] * W_time[td * FF + f];
    tfb[o] = acc;
  }
}

// ---------------------------------------------------------------------------
// Kernel B: per frame  h = relu( (Ahat @ X^T) @ Wgcn + tfb ),  h -> ws bf16
// 1 wave per frame, 4 waves per WG
// ---------------------------------------------------------------------------
__global__ __launch_bounds__(128)
void gcn_kernel(const float* x, unsigned char* ws) {
  __shared__ unsigned short sP[4][32 * 64];   // stage-1 result per wave
  __shared__ unsigned short sH[4][32 * 128];  // epilogue staging per wave

  int tid = threadIdx.x;
  int w = tid >> 5, l = tid & 31;
  int l15 = l & 15;
  int kb = (l < 16) ? 0 : 8;
  int frame = blockIdx.x * 4 + w;
  int n = frame >> 8, t = frame & 255;

  const unsigned short* Ahat = (const unsigned short*)(ws + WS_AHAT);
  const unsigned short* WgT  = (const unsigned short*)(ws + WS_WGT);
  const float*          tfb  = (const float*)(ws + WS_TFB);
  unsigned short*       Hout = (unsigned short*)(ws + WS_H);

  // ---- stage 1: P[32x64] = Ahat_pad[32x32] @ X^T[32x64] ----
  BfTile A1[2];
#pragma unroll
  for (int mt = 0; mt < 2; ++mt) {
    int row = mt * 16 + l15;
    A1[mt].q[0] = *(const uint4*)(Ahat + row * 32 + kb);
    A1[mt].q[1] = *(const uint4*)(Ahat + row * 32 + kb + 16);
  }
  BfTile B1[4];
#pragma unroll
  for (int nt = 0; nt < 4; ++nt) {
    int c = nt * 16 + l15;
    const float* xp = x + (((size_t)n * CIN + c) * TT + t) * VV;
#pragma unroll
    for (int j = 0; j < 4; ++j)   // v = kb+2j, kb+2j+1  (all < 16, in range)
      B1[nt].p[j] = pk2bf(xp[kb + 2 * j], xp[kb + 2 * j + 1]);
#pragma unroll
    for (int j = 0; j < 4; ++j) { // v = kb+16+2j (may exceed 24 -> zero pad)
      int v2 = kb + 16 + 2 * j;
      float a = (v2 < VV) ? xp[v2] : 0.0f;
      float b = (v2 + 1 < VV) ? xp[v2 + 1] : 0.0f;
      B1[nt].p[4 + j] = pk2bf(a, b);
    }
  }
  v8f P[2][4];
#pragma unroll
  for (int mt = 0; mt < 2; ++mt)
#pragma unroll
    for (int nt = 0; nt < 4; ++nt) {
      v8f z = {};
      P[mt][nt] = __builtin_amdgcn_wmma_f32_16x16x32_bf16(
          false, A1[mt].v, false, B1[nt].v, (short)0, z, false, false);
    }

  // D-layout -> LDS (bf16) so stage 2 can reload in A-layout
  unsigned short* sp = sP[w];
#pragma unroll
  for (int mt = 0; mt < 2; ++mt)
#pragma unroll
    for (int nt = 0; nt < 4; ++nt)
#pragma unroll
      for (int r = 0; r < 8; ++r) {
        int m = mt * 16 + (l >> 4) * 8 + r;
        int c = nt * 16 + l15;
        sp[m * 64 + c] = f2bf(P[mt][nt][r]);
      }

  // ---- stage 2: acc[32x128] = P[32x64] @ WgT^T (K = 64) ----
  v8f acc[2][8];
#pragma unroll
  for (int mt = 0; mt < 2; ++mt)
#pragma unroll
    for (int nt = 0; nt < 8; ++nt) acc[mt][nt] = (v8f){};

  BfTile A2[2][2];
#pragma unroll
  for (int mt = 0; mt < 2; ++mt)
#pragma unroll
    for (int kt = 0; kt < 2; ++kt) {
      int m = mt * 16 + l15;
      const unsigned short* p = sp + m * 64 + kt * 32 + kb;
      A2[mt][kt].q[0] = *(const uint4*)p;
      A2[mt][kt].q[1] = *(const uint4*)(p + 16);
    }
#pragma unroll
  for (int nt = 0; nt < 8; ++nt) {
    int f = nt * 16 + l15;
#pragma unroll
    for (int kt = 0; kt < 2; ++kt) {
      BfTile B2;
      const unsigned short* p = WgT + f * 64 + kt * 32 + kb;
      B2.q[0] = *(const uint4*)p;
      B2.q[1] = *(const uint4*)(p + 16);
      acc[0][nt] = __builtin_amdgcn_wmma_f32_16x16x32_bf16(
          false, A2[0][kt].v, false, B2.v, (short)0, acc[0][nt], false, false);
      acc[1][nt] = __builtin_amdgcn_wmma_f32_16x16x32_bf16(
          false, A2[1][kt].v, false, B2.v, (short)0, acc[1][nt], false, false);
    }
  }

  // ---- epilogue: + tfb, relu, bf16, coalesced store h[n][v][t][f] ----
  unsigned short* sh = sH[w];
#pragma unroll
  for (int nt = 0; nt < 8; ++nt) {
    int f = nt * 16 + l15;
    float bias = tfb[n * FF + f];
#pragma unroll
    for (int mt = 0; mt < 2; ++mt)
#pragma unroll
      for (int r = 0; r < 8; ++r) {
        int m = mt * 16 + (l >> 4) * 8 + r;
        float val = acc[mt][nt][r] + bias;
        sh[m * FF + f] = f2bf(val > 0.0f ? val : 0.0f);
      }
  }
  for (int chunk = l; chunk < VV * 16; chunk += 32) {
    int v = chunk >> 4, c = chunk & 15;
    uint4 d = *(const uint4*)(sh + v * FF + c * 8);
    *(uint4*)(Hout + (((size_t)n * VV + v) * TT + t) * FF + c * 8) = d;
  }
}

// ---------------------------------------------------------------------------
// Kernel C: TCN  out[n,:,:,v] = relu( sum_k H_shift_k @ Wk^T + b_tcn )
// one WG per (n,v); 8 waves, each owns 32 t-rows x 128 f_out
// per-tap 32KB weight tile staged into LDS via the Tensor Data Mover
// ---------------------------------------------------------------------------
__global__ __launch_bounds__(256)
void tcn_kernel(const unsigned char* ws, const float* b_tcn, float* out) {
  __shared__ unsigned short sWk[FF * FF];   // 32 KB per-tap weights

  int tid = threadIdx.x;
  int w = tid >> 5, l = tid & 31;
  int l15 = l & 15;
  int kb = (l < 16) ? 0 : 8;
  int n = blockIdx.x / VV, v = blockIdx.x % VV;

  const unsigned short* Hbase =
      (const unsigned short*)(ws + WS_H) + ((size_t)n * VV + v) * TT * FF;
  const unsigned short* WkT = (const unsigned short*)(ws + WS_WKT);

  // pull this wave's 32 h-rows toward the WGP (re-read for 3 taps)
  __builtin_prefetch(Hbase + (w * 32 + l) * FF, 0, 0);
  __builtin_prefetch(Hbase + (w * 32 + l) * FF + 64, 0, 0);

  v8f acc[2][8];
#pragma unroll
  for (int mt = 0; mt < 2; ++mt)
#pragma unroll
    for (int nt = 0; nt < 8; ++nt) acc[mt][nt] = (v8f){};

  for (int k = 0; k < 3; ++k) {
    __syncthreads();   // previous tap's sWk fully consumed
#ifdef USE_TDM
    if (w == 0) {
      // Tensor DMA: 2-D descriptor, 128x128 bf16 tile, global -> LDS
      typedef __attribute__((ext_vector_type(4))) unsigned tdm_v4u;
      typedef __attribute__((ext_vector_type(8))) int      tdm_v8i;
      typedef __attribute__((ext_vector_type(4))) int      tdm_v4i;
      unsigned long long ga =
          (unsigned long long)(const void*)(WkT + k * FF * FF);
      unsigned lds_off = (unsigned)(unsigned long long)(void*)sWk;
      tdm_v4u g0;
      g0[0] = 1u;                                    // count = 1 descriptor
      g0[1] = lds_off;                               // lds_addr (bytes)
      g0[2] = (unsigned)ga;                          // global_addr[31:0]
      g0[3] = (unsigned)((ga >> 32) & 0x01FFFFFFu)   // global_addr[56:32]
              | (2u << 30);                          // type = 2 ("image")
      tdm_v8i g1;
      g1[0] = (int)(1u << 16);     // wg_mask=0, data_size=1 (2 bytes)
      g1[1] = (int)(128u << 16);   // tensor_dim0 = 128 (bits 79:48 low half)
      g1[2] = (int)(128u << 16);   // tensor_dim1 = 128 (bits 95:80)
      g1[3] = (int)(128u << 16);   // tile_dim0 = 128   (bits 127:112)
      g1[4] = 128;                 // tile_dim1 = 128, tile_dim2 = 0
      g1[5] = 128;                 // tensor_dim0_stride = 128
      g1[6] = 0;
      g1[7] = 0;
      tdm_v4i gz = {0, 0, 0, 0};
#if __has_include(<hip/amd_detail/amd_gfx1250_TDM.h>)
      __builtin_amdgcn_tensor_load_to_lds(g0, g1, gz, gz,
                                          (tdm_v8i){0, 0, 0, 0, 0, 0, 0, 0}, 0);
#else
      __builtin_amdgcn_tensor_load_to_lds(g0, g1, gz, gz, 0);
#endif
      __builtin_amdgcn_s_wait_tensorcnt(0);
    }
    __syncthreads();
#else
    {
      const uint4* src = (const uint4*)(WkT + k * FF * FF);
      uint4* dst = (uint4*)sWk;
      for (int i = tid; i < FF * FF / 8; i += 256) dst[i] = src[i];
    }
    __syncthreads();
#endif

    // A tiles: rows t + k - 1 (zero outside [0,256))
    BfTile A[2][4];
#pragma unroll
    for (int mt = 0; mt < 2; ++mt)
#pragma unroll
      for (int kt = 0; kt < 4; ++kt) {
        int trow = w * 32 + mt * 16 + l15 + k - 1;
        if ((unsigned)trow < (unsigned)TT) {
          const unsigned short* p = Hbase + trow * FF + kt * 32 + kb;
          A[mt][kt].q[0] = *(const uint4*)p;
          A[mt][kt].q[1] = *(const uint4*)(p + 16);
        } else {
          A[mt][kt].q[0] = make_uint4(0, 0, 0, 0);
          A[mt][kt].q[1] = make_uint4(0, 0, 0, 0);
        }
      }

#pragma unroll
    for (int nt = 0; nt < 8; ++nt) {
      int fo = nt * 16 + l15;
#pragma unroll
      for (int kt = 0; kt < 4; ++kt) {
        BfTile B;
        const unsigned short* p = sWk + fo * FF + kt * 32 + kb;
        B.q[0] = *(const uint4*)p;
        B.q[1] = *(const uint4*)(p + 16);
        acc[0][nt] = __builtin_amdgcn_wmma_f32_16x16x32_bf16(
            false, A[0][kt].v, false, B.v, (short)0, acc[0][nt], false, false);
        acc[1][nt] = __builtin_amdgcn_wmma_f32_16x16x32_bf16(
            false, A[1][kt].v, false, B.v, (short)0, acc[1][nt], false, false);
      }
    }
  }

  // epilogue: + b_tcn, relu, store fp32 out[n][fo][t][v] (L2 merges lines)
#pragma unroll
  for (int nt = 0; nt < 8; ++nt) {
    int fo = nt * 16 + l15;
    float bias = b_tcn[fo];
#pragma unroll
    for (int mt = 0; mt < 2; ++mt)
#pragma unroll
      for (int r = 0; r < 8; ++r) {
        int t = w * 32 + mt * 16 + (l >> 4) * 8 + r;
        float val = acc[mt][nt][r] + bias;
        out[(((size_t)n * FF + fo) * TT + t) * VV + v] = val > 0.0f ? val : 0.0f;
      }
  }
}

// ---------------------------------------------------------------------------
extern "C" void kernel_launch(void* const* d_in, const int* in_sizes, int n_in,
                              void* d_out, int out_size, void* d_ws, size_t ws_size,
                              hipStream_t stream) {
  const float* x      = (const float*)d_in[0];
  const int*   ei     = (const int*)d_in[1];
  const float* t_emb  = (const float*)d_in[2];
  const float* W_gcn  = (const float*)d_in[3];
  const float* b_gcn  = (const float*)d_in[4];
  const float* W_time = (const float*)d_in[5];
  const float* b_time = (const float*)d_in[6];
  const float* W_tcn  = (const float*)d_in[7];
  const float* b_tcn  = (const float*)d_in[8];
  unsigned char* ws   = (unsigned char*)d_ws;

  prep_kernel<<<1, 256, 0, stream>>>(ei, t_emb, W_gcn, b_gcn, W_time, b_time,
                                     W_tcn, ws);
  gcn_kernel<<<(NB * TT) / 4, 128, 0, stream>>>(x, ws);
  tcn_kernel<<<NB * VV, 256, 0, stream>>>(ws, b_tcn, (float*)d_out);
}
